// RegionalAttention_68264210203180
// MI455X (gfx1250) — compile-verified, matrix operand order checked by reading, not run on hardware
//
#include <hip/hip_runtime.h>

typedef _Float16 half_t;
typedef _Float16 v16h __attribute__((ext_vector_type(16)));
typedef _Float16 v8h  __attribute__((ext_vector_type(8)));
typedef _Float16 v4h  __attribute__((ext_vector_type(4)));
typedef _Float16 v2h  __attribute__((ext_vector_type(2)));
typedef float    v8f  __attribute__((ext_vector_type(8)));
typedef __attribute__((address_space(3))) half_t lds_half;

#define LDIM   512
#define LL     (512 * 512)
#define CH     64
#define HEADS  4
#define HDIM   16
#define TP     128          // pass-1: pixels (one row strip) per block
#define PSTR   72           // halves per pixel row: 144B = 9x16B, b128-aligned
#define WSTR   72           // halves per weight row
#define NP     16           // pass-2 tile side (16x16 pixels)
#define HS     18           // halo side
#define QK_SCALE 0.125f     // 1/sqrt(64)

__device__ __forceinline__ v16h frag_cat(v8h lo, v8h hi) {
    return __builtin_shufflevector(lo, hi, 0, 1, 2, 3, 4, 5, 6, 7,
                                           8, 9, 10, 11, 12, 13, 14, 15);
}

// async global->LDS copy of one 32B vector (2 x b128; offset applies to both
// the LDS and global address per the ISA async pseudocode). ASYNCcnt-tracked.
__device__ __forceinline__ void async_copy32(half_t* ldst, const half_t* gsrc) {
    unsigned ldsoff = (unsigned)(size_t)(lds_half*)ldst;
    asm volatile("global_load_async_to_lds_b128 %0, %1, off\n\t"
                 "global_load_async_to_lds_b128 %0, %1, off offset:16"
                 :: "v"(ldsoff), "v"((const void*)gsrc) : "memory");
}
__device__ __forceinline__ void wait_async0() {
    asm volatile("s_wait_asynccnt 0" ::: "memory");
}

// ---------------------------------------------------------------------------
// Pass 1: fused Q/K/V projection, D = W (16x64) x X^T (64x16) per tile.
// A = nn.Linear weight rows as-is, B = pixel-major x tile (both contiguous
// 16B fragment runs -> ds_load_b128).  q/k/v: [head][y][x][dim] f16.
// ---------------------------------------------------------------------------
__global__ __launch_bounds__(256) void qkv_proj_kernel(
    const float* __restrict__ x,
    const float* __restrict__ Wq, const float* __restrict__ Wk,
    const float* __restrict__ Wv,
    half_t* __restrict__ qws, half_t* __restrict__ kws, half_t* __restrict__ vws)
{
    __shared__ __align__(16) half_t ldsX[TP * PSTR];
    __shared__ __align__(16) half_t ldsW[3][CH * WSTR];

    const int t  = threadIdx.x;
    const int b  = blockIdx.x;
    const int y  = b >> 2;
    const int x0 = (b & 3) * TP;

    {   // stage x tile: coalesced b32 reads, packed h2 LDS stores
        const int p = t & 127, chalf = t >> 7;
        const size_t rowoff = (size_t)y * LDIM + x0 + p;
        #pragma unroll
        for (int i = 0; i < 16; ++i) {
            int c = chalf * 32 + 2 * i;
            float v0 = x[(size_t)c * LL + rowoff];
            float v1 = x[(size_t)(c + 1) * LL + rowoff];
            v2h pk = {(half_t)v0, (half_t)v1};
            *(v2h*)&ldsX[p * PSTR + c] = pk;
        }
        __builtin_prefetch(&x[rowoff + (y + 1 < LDIM ? LDIM : 0)], 0, 0);
    }
    #pragma unroll
    for (int m = 0; m < 3; ++m) {       // stage weights row-major, f32->f16
        const float2* W2 = (const float2*)((m == 0) ? Wq : (m == 1) ? Wk : Wv);
        #pragma unroll
        for (int i = 0; i < 8; ++i) {
            int id = t + 256 * i;
            int o = id >> 5, cp = id & 31;
            float2 wv = W2[o * 32 + cp];
            v2h pk = {(half_t)wv.x, (half_t)wv.y};
            *(v2h*)&ldsW[m][o * WSTR + 2 * cp] = pk;
        }
    }
    __syncthreads();

    const int lane = t & 31, wave = t >> 5;
    const int pixbase = wave * 16;
    const int nn = lane & 15, hi = lane >> 4;

    const v8h* brow = (const v8h*)&ldsX[(pixbase + nn) * PSTR];
    const v16h bf0 = frag_cat(brow[hi],     brow[2 + hi]);   // K = 0..31
    const v16h bf1 = frag_cat(brow[4 + hi], brow[6 + hi]);   // K = 32..63

    const int xg = x0 + pixbase + nn;

    #pragma unroll
    for (int m = 0; m < 3; ++m) {
        half_t* dstw = (m == 0) ? qws : (m == 1) ? kws : vws;
        #pragma unroll
        for (int mt = 0; mt < HEADS; ++mt) {
            const v8h* arow = (const v8h*)&ldsW[m][(mt * 16 + nn) * WSTR];
            v16h af0 = frag_cat(arow[hi],     arow[2 + hi]);
            v16h af1 = frag_cat(arow[4 + hi], arow[6 + hi]);
            v8f acc = {};
            acc = __builtin_amdgcn_wmma_f32_16x16x32_f16(false, af0, false, bf0,
                                                         (short)0, acc, false, false);
            acc = __builtin_amdgcn_wmma_f32_16x16x32_f16(false, af1, false, bf1,
                                                         (short)0, acc, false, false);
            v8h hv = __builtin_convertvector(acc, v8h);
            size_t off = (((size_t)mt * LDIM + y) * LDIM + xg) * HDIM + hi * 8;
            *(v8h*)&dstw[off] = hv;   // coalesced b128: dims 8hi..8hi+7, head mt
        }
    }
}

// ---------------------------------------------------------------------------
// Pass 2: 16x16 pixel tile, 18x18 zero-filled k/v halo staged into LDS with
// async global->LDS b128 copies (ASYNCcnt), packed-f16 attention math, then
// the 64x64 output projection via WMMA.  Zero halo reproduces the reference
// zero-pad semantics (OOB score == 0 enters the softmax) branch-free.
// ---------------------------------------------------------------------------
__global__ __launch_bounds__(256) void attn_proj_kernel(
    const half_t* __restrict__ qws, const half_t* __restrict__ kws,
    const half_t* __restrict__ vws,
    const float* __restrict__ Wo, const float* __restrict__ bo,
    float* __restrict__ out)
{
    __shared__ __align__(16) half_t ldsK[HEADS * HS * HS * HDIM];  // 41.5 KB
    __shared__ __align__(16) half_t ldsV[HEADS * HS * HS * HDIM];  // 41.5 KB
    __shared__ __align__(16) half_t ldsA[NP * NP * PSTR];          // 36.9 KB
    __shared__ __align__(16) half_t ldsWo[CH * WSTR];              //  9.2 KB
    __shared__ __align__(32) float  ldsBias[CH];

    const int t  = threadIdx.x;
    const int b  = blockIdx.x;
    const int y0 = (b >> 5) * NP;
    const int x0 = (b & 31) * NP;

    // ---- stage k/v halo via async copies (zeros for OOB entries)
    #pragma unroll
    for (int m = 0; m < 2; ++m) {
        const half_t* src = (m == 0) ? kws : vws;
        half_t* dst = (m == 0) ? ldsK : ldsV;
        #pragma unroll
        for (int i = 0; i < 6; ++i) {            // 1296 vectors / 256 threads
            int v = t + i * 256;
            if (v < HEADS * HS * HS) {
                int head = v / (HS * HS);
                int rem  = v - head * (HS * HS);
                int ly = rem / HS, lx = rem - ly * HS;
                int gy = y0 - 1 + ly, gx = x0 - 1 + lx;
                half_t* dp = dst + v * HDIM;
                if (gy >= 0 && gy < LDIM && gx >= 0 && gx < LDIM) {
                    async_copy32(dp, src +
                        (((size_t)head * LDIM + gy) * LDIM + gx) * HDIM);
                } else {
                    v8h z = {};
                    ((v8h*)dp)[0] = z; ((v8h*)dp)[1] = z;
                }
            }
        }
    }
    {   // stage Wo (row-major) + bias, overlapped with async transfers
        const float2* W2 = (const float2*)Wo;
        #pragma unroll
        for (int i = 0; i < 8; ++i) {
            int id = t + 256 * i;
            int o = id >> 5, cp = id & 31;
            float2 wv = W2[o * 32 + cp];
            v2h pk = {(half_t)wv.x, (half_t)wv.y};
            *(v2h*)&ldsWo[o * WSTR + 2 * cp] = pk;
        }
        if (t < CH) ldsBias[t] = bo[t];
    }
    wait_async0();
    __syncthreads();

    // ---- attention: 256 px x 4 heads = 1024 tasks, 4 per thread,
    //      packed f16 math (v_pk_fma_f16), branch-free neighbor loop
    #pragma unroll
    for (int it = 0; it < 4; ++it) {
        int id   = t + it * 256;
        int head = id >> 8;
        int pi   = id & 255;
        int py = pi >> 4, px = pi & 15;

        const v8h* qp = (const v8h*)(qws +
            (((size_t)head * LDIM + (y0 + py)) * LDIM + (x0 + px)) * HDIM);
        v8h q0 = qp[0], q1 = qp[1];

        const half_t* kbase = ldsK + ((head * HS + py) * HS + px) * HDIM;
        const half_t* vbase = ldsV + ((head * HS + py) * HS + px) * HDIM;

        float sc[9];
        #pragma unroll
        for (int i = 0; i < 3; ++i)
            #pragma unroll
            for (int j = 0; j < 3; ++j) {
                const v8h* kp = (const v8h*)(kbase + (i * HS + j) * HDIM);
                v8h pr = q0 * kp[0] + q1 * kp[1];
                v4h r4 = __builtin_shufflevector(pr, pr, 0, 1, 2, 3)
                       + __builtin_shufflevector(pr, pr, 4, 5, 6, 7);
                v2h r2 = __builtin_shufflevector(r4, r4, 0, 1)
                       + __builtin_shufflevector(r4, r4, 2, 3);
                sc[i * 3 + j] = ((float)r2[0] + (float)r2[1]) * QK_SCALE;
            }

        float mx = sc[0];
        #pragma unroll
        for (int w = 1; w < 9; ++w) mx = fmaxf(mx, sc[w]);
        float sum = 0.f;
        #pragma unroll
        for (int w = 0; w < 9; ++w) { sc[w] = __expf(sc[w] - mx); sum += sc[w]; }
        float inv = 1.f / sum;

        v8h o0 = {}, o1 = {};
        #pragma unroll
        for (int i = 0; i < 3; ++i)
            #pragma unroll
            for (int j = 0; j < 3; ++j) {
                const v8h* vp = (const v8h*)(vbase + (i * HS + j) * HDIM);
                half_t ah = (half_t)(sc[i * 3 + j] * inv);
                v8h av = {ah, ah, ah, ah, ah, ah, ah, ah};
                o0 += av * vp[0];
                o1 += av * vp[1];
            }
        *(v8h*)&ldsA[pi * PSTR + head * HDIM]     = o0;
        *(v8h*)&ldsA[pi * PSTR + head * HDIM + 8] = o1;
    }
    __syncthreads();

    // ---- output projection: D = Wo (16x64) x attn^T (64x16); 16 N-tiles
    //      (= image rows of the tile), 2 per wave
    const int lane = t & 31, wave = t >> 5;
    const int nn = lane & 15, hi = lane >> 4;

    #pragma unroll
    for (int nt2 = 0; nt2 < 2; ++nt2) {
        int tile = wave * 2 + nt2;              // 0..15 == tile row py
        int pi = tile * 16 + nn;                // pixel: py=tile, px=nn
        const v8h* brow = (const v8h*)&ldsA[pi * PSTR];
        v16h bf0 = frag_cat(brow[hi],     brow[2 + hi]);
        v16h bf1 = frag_cat(brow[4 + hi], brow[6 + hi]);

        const size_t outrow = (size_t)(y0 + tile) * LDIM + x0 + nn;
        #pragma unroll
        for (int mt = 0; mt < 4; ++mt) {
            const v8h* arow = (const v8h*)&ldsWo[(mt * 16 + nn) * WSTR];
            v16h af0 = frag_cat(arow[hi],     arow[2 + hi]);
            v16h af1 = frag_cat(arow[4 + hi], arow[6 + hi]);
            v8f acc = {};
            acc = __builtin_amdgcn_wmma_f32_16x16x32_f16(false, af0, false, bf0,
                                                         (short)0, acc, false, false);
            acc = __builtin_amdgcn_wmma_f32_16x16x32_f16(false, af1, false, bf1,
                                                         (short)0, acc, false, false);
            acc += *(const v8f*)&ldsBias[mt * 16 + hi * 8];
            #pragma unroll
            for (int r = 0; r < 8; ++r) {       // 64B runs per channel row
                int c = mt * 16 + 8 * hi + r;
                out[(size_t)c * LL + outrow] = acc[r];
            }
        }
    }
}

// ---------------------------------------------------------------------------
extern "C" void kernel_launch(void* const* d_in, const int* in_sizes, int n_in,
                              void* d_out, int out_size, void* d_ws, size_t ws_size,
                              hipStream_t stream) {
    (void)in_sizes; (void)n_in; (void)out_size; (void)ws_size;
    const float* x  = (const float*)d_in[0];
    const float* Wq = (const float*)d_in[1];
    const float* Wk = (const float*)d_in[2];
    const float* Wv = (const float*)d_in[3];
    const float* Wo = (const float*)d_in[4];
    const float* bo = (const float*)d_in[5];
    float* out = (float*)d_out;

    const size_t qkv_elems = (size_t)HEADS * LDIM * LDIM * HDIM; // 32MB each
    half_t* qws = (half_t*)d_ws;
    half_t* kws = qws + qkv_elems;
    half_t* vws = kws + qkv_elems;

    qkv_proj_kernel<<<dim3(LDIM * (LDIM / TP)), 256, 0, stream>>>(
        x, Wq, Wk, Wv, qws, kws, vws);
    attn_proj_kernel<<<dim3((LDIM / NP) * (LDIM / NP)), 256, 0, stream>>>(
        qws, kws, vws, Wo, bo, out);
}
